// Decoder_48438641164932
// MI455X (gfx1250) — matrix operand order repaired and checked by reading, not compiled, not run on hardware
//
#include <hip/hip_runtime.h>
#include <hip/hip_bf16.h>
#include <math.h>

#define B_DIM 8
#define T_DIM 128
#define S_DIM 400
#define D_DIM 512
#define V_DIM 50000
#define M_DIM (B_DIM * T_DIM)   // 1024
#define EPS_LN 1e-6f
#define KSTEP 32
#define NTILE 256                // block N tile (rows of W)
#define TROW 40                  // LDS halves per tile row: 32 data + 8 pad (80 B)

typedef __attribute__((ext_vector_type(16))) _Float16 v16h;
typedef __attribute__((ext_vector_type(8)))  _Float16 v8h;
typedef __attribute__((ext_vector_type(4)))  _Float16 v4h;
typedef __attribute__((ext_vector_type(8)))  float    v8f;
typedef __attribute__((ext_vector_type(4)))  float    v4f;
typedef __attribute__((ext_vector_type(4)))  unsigned int uint32x4;
typedef __attribute__((ext_vector_type(8)))  int int32x8;
typedef __attribute__((ext_vector_type(4)))  int int32x4;

union HFrag { v16h v; v8h h[2]; };
union H4    { v4h v; _Float16 e[4]; };

#if __has_builtin(__builtin_amdgcn_tensor_load_to_lds) && \
    __has_builtin(__builtin_amdgcn_s_wait_tensorcnt)
#define HAVE_TDM 1
#else
#define HAVE_TDM 0
#endif

// ---------------- reductions (wave32) ----------------
__device__ __forceinline__ float warpSum(float v) {
  #pragma unroll
  for (int off = 16; off > 0; off >>= 1) v += __shfl_down(v, off, 32);
  return v;
}
__device__ __forceinline__ float warpMax(float v) {
  #pragma unroll
  for (int off = 16; off > 0; off >>= 1) v = fmaxf(v, __shfl_down(v, off, 32));
  return v;
}
__device__ __forceinline__ float blockSum(float v) {
  __shared__ float s[8];
  __shared__ float r;
  const int lane = threadIdx.x & 31, w = threadIdx.x >> 5;
  v = warpSum(v);
  if (lane == 0) s[w] = v;
  __syncthreads();
  if (w == 0) {
    float t = (lane < 8) ? s[lane] : 0.0f;
    t = warpSum(t);
    if (lane == 0) r = t;
  }
  __syncthreads();
  float out = r;
  __syncthreads();
  return out;
}
__device__ __forceinline__ float blockMax(float v) {
  __shared__ float s[8];
  __shared__ float r;
  const int lane = threadIdx.x & 31, w = threadIdx.x >> 5;
  v = warpMax(v);
  if (lane == 0) s[w] = v;
  __syncthreads();
  if (w == 0) {
    float t = (lane < 8) ? s[lane] : -3.4e38f;
    t = warpMax(t);
    if (lane == 0) r = t;
  }
  __syncthreads();
  float out = r;
  __syncthreads();
  return out;
}

// ---------------- 1) proj_w f32 -> f16 ----------------
__global__ __launch_bounds__(256) void prep_w(const float* __restrict__ w,
                                              _Float16* __restrict__ wh) {
  size_t i = ((size_t)blockIdx.x * 256 + threadIdx.x) * 4;
  if (i + 3 >= (size_t)V_DIM * D_DIM) return;
  v4f f = *(const v4f*)(w + i);
  H4 h;
  h.e[0] = (_Float16)f.x; h.e[1] = (_Float16)f.y;
  h.e[2] = (_Float16)f.z; h.e[3] = (_Float16)f.w;
  *(v4h*)(wh + i) = h.v;
}

// ---------------- 2) layernorm(x) -> f16 ----------------
__global__ __launch_bounds__(256) void prep_x(const float* __restrict__ x,
                                              const float* __restrict__ na,
                                              const float* __restrict__ nb,
                                              _Float16* __restrict__ xn) {
  const int row = blockIdx.x;
  const float* xr = x + (size_t)row * D_DIM;
  const int t0 = threadIdx.x, t1 = threadIdx.x + 256;
  float a = xr[t0], b = xr[t1];
  float mu = blockSum(a + b) * (1.0f / D_DIM);
  float da = a - mu, db = b - mu;
  float var = blockSum(da * da + db * db) * (1.0f / D_DIM);
  float inv = 1.0f / (sqrtf(var) + EPS_LN);
  _Float16* xo = xn + (size_t)row * D_DIM;
  xo[t0] = (_Float16)(na[t0] * da * inv + nb[t0]);
  xo[t1] = (_Float16)(na[t1] * db * inv + nb[t1]);
}

// ---------------- 3) m_h[b,s] = wh . memory[b,s,:] ----------------
__global__ __launch_bounds__(256) void mh_kernel(const float* __restrict__ mem,
                                                 const float* __restrict__ whw,
                                                 float* __restrict__ mh) {
  const int bs = blockIdx.x;  // b*S + s
  const float* m = mem + (size_t)bs * D_DIM;
  const int t0 = threadIdx.x, t1 = threadIdx.x + 256;
  float v = m[t0] * whw[t0] + m[t1] * whw[t1];
  v = blockSum(v);
  if (threadIdx.x == 0) mh[bs] = v;
}

// ---------------- 4) p_gen ----------------
__global__ __launch_bounds__(256) void logits_kernel(
    const float* __restrict__ attn, const float* __restrict__ mh,
    const float* __restrict__ x, const float* __restrict__ xt,
    const float* __restrict__ wsw, const float* __restrict__ wxw,
    const float* __restrict__ whb, const float* __restrict__ wsb,
    const float* __restrict__ wxb, const float* __restrict__ bptr,
    float* __restrict__ pgen) {
  const int row = blockIdx.x;           // b*T + t
  const int b = row / T_DIM;
  float acc = 0.0f;
  const float* ar = attn + (size_t)row * S_DIM;
  const float* mr = mh + (size_t)b * S_DIM;
  for (int s = threadIdx.x; s < S_DIM; s += 256) acc += ar[s] * mr[s];
  const float* xr = x + (size_t)row * D_DIM;
  const float* xtr = xt + (size_t)row * D_DIM;
  const int t0 = threadIdx.x, t1 = threadIdx.x + 256;
  acc += xr[t0] * wsw[t0] + xr[t1] * wsw[t1];
  acc += xtr[t0] * wxw[t0] + xtr[t1] * wxw[t1];
  acc = blockSum(acc);
  if (threadIdx.x == 0) {
    float logit = acc + whb[0] + wsb[0] + wxb[0] + bptr[0];
    pgen[row] = 1.0f / (1.0f + __expf(-logit));
  }
}

// ---------------- TDM descriptor issue (ISA 08_async_tensor §8) ----------------
#if HAVE_TDM
__device__ __forceinline__ void tdm_load_tile(unsigned int lds_addr,
                                              const _Float16* gsrc, int rows) {
  const unsigned long long ga = (unsigned long long)(uintptr_t)gsrc;
  uint32x4 g0;
  g0.x = 1u;                                   // count=1 valid descriptor
  g0.y = lds_addr;                             // lds_addr [63:32]
  g0.z = (unsigned int)ga;                     // global_addr[31:0]
  g0.w = (unsigned int)(ga >> 32) | (2u << 30);// global_addr[56:32] | type=2
  int32x8 g1;
  // data_size=2B (code 1), pad_enable, pad_interval=16 DWORDs (code 3),
  // pad_amount=4 DWORDs (code 3): row 64B -> 80B in LDS
  g1[0] = (int)((1u << 16) | (1u << 20) | (3u << 22) | (3u << 25));
  g1[1] = (int)(((unsigned)D_DIM & 0xFFFFu) << 16);            // tensor_dim0 lo
  g1[2] = (int)((((unsigned)D_DIM) >> 16) |
                (((unsigned)V_DIM & 0xFFFFu) << 16));          // dim0 hi | dim1 lo
  g1[3] = (int)((((unsigned)V_DIM) >> 16) |
                ((unsigned)KSTEP << 16));                      // dim1 hi | tile_dim0
  g1[4] = rows;                                                // tile_dim1 | tile_dim2=0
  g1[5] = D_DIM;                                               // tensor_dim0_stride lo
  g1[6] = 0;
  g1[7] = 0;
  int32x4 gz4 = {0, 0, 0, 0};
  int32x8 gz8 = {0, 0, 0, 0, 0, 0, 0, 0};
  // 6-arg flavor (clang-23 / therock-10.0 headers)
  __builtin_amdgcn_tensor_load_to_lds(g0, g1, gz4, gz4, gz8, 0);
}
#endif

// ---------------- 5) WMMA GEMM: dec = Xn @ Wh^T + proj_b ----------------
// 256 threads = 8 waves (2M x 4N). Wave tile 16M x 64N, block tile 32M x 256N.
// W block-tile staged in LDS (TDM double-buffered), A frags from global (L0-hot).
__global__ __launch_bounds__(256) void gemm_wmma(
    const _Float16* __restrict__ Xn, const _Float16* __restrict__ Wh,
    const float* __restrict__ pb, float* __restrict__ out) {
  __shared__ _Float16 tile[2][NTILE * TROW];   // 2 x 20 KB
  const int lane = threadIdx.x & 31;
  const int wave = threadIdx.x >> 5;
  const int l16 = lane & 15;
  const int lhi = lane >> 4;
  const int waveN = wave & 3;
  const int m0 = blockIdx.y * 32 + (wave >> 2) * 16;
  const int n0b = blockIdx.x * NTILE;

  // real zero-init stores so LDS reads can't be folded to undef
  {
    v8h z = {};
    v8h* t8 = (v8h*)&tile[0][0];
    for (int i = threadIdx.x; i < 2 * NTILE * TROW / 8; i += 256) t8[i] = z;
  }
  __syncthreads();

  int rows = V_DIM - n0b; if (rows > NTILE) rows = NTILE;
  const _Float16* wbase = Wh + (size_t)n0b * D_DIM;

#if HAVE_TDM
  const unsigned int lds0 = (unsigned int)(uintptr_t)&tile[0][0];
  const unsigned int lds1 = (unsigned int)(uintptr_t)&tile[1][0];
  if (wave == 0) tdm_load_tile(lds0, wbase, rows);
#endif

  v8f acc[4] = {};
  const _Float16* arow = Xn + (size_t)(m0 + l16) * D_DIM;
  int buf = 0;

  for (int k0 = 0; k0 < D_DIM; k0 += KSTEP) {
#if HAVE_TDM
    if (wave == 0) __builtin_amdgcn_s_wait_tensorcnt(0);
    __syncthreads();                           // publish current buffer
    if (wave == 0 && (k0 + KSTEP) < D_DIM)
      tdm_load_tile(buf ? lds0 : lds1, wbase + (k0 + KSTEP), rows);
#else
    // cooperative fallback fill: thread t copies row t's 32 halves
    {
      const int r = threadIdx.x;               // NTILE == 256 == blockDim
      const _Float16* gsrc = wbase + (size_t)((r < rows) ? r : 0) * D_DIM + k0;
      v8h lo = *(const v8h*)gsrc;
      v8h hi = *(const v8h*)(gsrc + 8);
      v8h lo2 = *(const v8h*)(gsrc + 16);
      v8h hi2 = *(const v8h*)(gsrc + 24);
      _Float16* drow = &tile[buf][r * TROW];
      *(v8h*)(drow) = lo;  *(v8h*)(drow + 8)  = hi;
      *(v8h*)(drow + 16) = lo2; *(v8h*)(drow + 24) = hi2;
    }
    __syncthreads();
#endif
    // A fragment (16x32 f16), ISA 7.12.2 layout
    HFrag a;
    a.h[0] = *(const v8h*)(arow + k0 + 8 * lhi);
    a.h[1] = *(const v8h*)(arow + k0 + 16 + 8 * lhi);
    __builtin_prefetch(arow + k0 + KSTEP, 0, 1);

    const _Float16* tb = &tile[buf][0];
    #pragma unroll
    for (int j = 0; j < 4; ++j) {
      const int rl = waveN * 64 + j * 16 + l16;  // local W row (B column)
      const _Float16* bp = tb + rl * TROW + 8 * lhi;
      HFrag bf;
      bf.h[0] = *(const v8h*)(bp);
      bf.h[1] = *(const v8h*)(bp + 16);
      acc[j] = __builtin_amdgcn_wmma_f32_16x16x32_f16(
          false, a.v, false, bf.v, (short)0, acc[j], false, false);
    }
    __syncthreads();                           // all reads of buf done
    buf ^= 1;
  }

  // D layout: VGPR r -> M=r (lanes 0-15) / M=r+8 (lanes 16-31); N = lane%16
  #pragma unroll
  for (int j = 0; j < 4; ++j) {
    const int col = n0b + waveN * 64 + j * 16 + l16;
    if (col >= V_DIM) continue;
    const float bias = pb[col];
    #pragma unroll
    for (int r = 0; r < 8; ++r) {
      const int row = m0 + r + 8 * lhi;
      out[(size_t)row * V_DIM + col] = acc[j][r] + bias;
    }
  }
}

// ---------------- 6) finalize: scatter + softmax + log, in place ----------------
__global__ __launch_bounds__(256) void finalize_kernel(
    const float* __restrict__ attn, const int* __restrict__ src,
    float* __restrict__ out, const float* __restrict__ pgen) {
  extern __shared__ float enc[];            // V_DIM floats = 200 KB LDS
  const int row = blockIdx.x;               // b*T + t
  const int b = row / T_DIM;
  float* z = out + (size_t)row * V_DIM;

  for (int v = threadIdx.x; v < V_DIM; v += 256) enc[v] = 0.0f;
  __syncthreads();
  const float* ar = attn + (size_t)row * S_DIM;
  const int* sr = src + (size_t)b * S_DIM;
  for (int s = threadIdx.x; s < S_DIM; s += 256)
    atomicAdd(&enc[sr[s]], ar[s]);          // ds_add_f32
  __syncthreads();

  float mx = -3.4e38f;
  for (int v = threadIdx.x; v < V_DIM; v += 256) mx = fmaxf(mx, z[v]);
  mx = blockMax(mx);

  float se = 0.0f;
  for (int v = threadIdx.x; v < V_DIM; v += 256) se += __expf(z[v] - mx);
  se = blockSum(se);
  const float inv = 1.0f / se;

  const float pg = pgen[row];
  const float omp = 1.0f - pg;
  for (int v = threadIdx.x; v < V_DIM; v += 256) {
    float sm = __expf(z[v] - mx) * inv;
    z[v] = __logf(pg * sm + omp * enc[v] + 1e-12f);
  }
}

// ---------------- launch ----------------
extern "C" void kernel_launch(void* const* d_in, const int* in_sizes, int n_in,
                              void* d_out, int out_size, void* d_ws, size_t ws_size,
                              hipStream_t stream) {
  const float* x      = (const float*)d_in[0];
  const float* x_t    = (const float*)d_in[1];
  const float* memory = (const float*)d_in[2];
  const float* attn   = (const float*)d_in[3];
  const int*   src    = (const int*)d_in[4];
  const float* norm_a = (const float*)d_in[5];
  const float* norm_b = (const float*)d_in[6];
  const float* proj_w = (const float*)d_in[7];
  const float* proj_b = (const float*)d_in[8];
  const float* wh_w   = (const float*)d_in[9];
  const float* wh_b   = (const float*)d_in[10];
  const float* ws_w   = (const float*)d_in[11];
  const float* ws_b   = (const float*)d_in[12];
  const float* wx_w   = (const float*)d_in[13];
  const float* wx_b   = (const float*)d_in[14];
  const float* bptr   = (const float*)d_in[15];

  float* out  = (float*)d_out;
  float* pgen = out + (size_t)M_DIM * V_DIM;   // p_gen region after 'out'

  char* ws = (char*)d_ws;
  _Float16* Wh = (_Float16*)ws;                                       // V*D f16
  _Float16* Xn = (_Float16*)(ws + (size_t)V_DIM * D_DIM * 2);         // M*D f16
  float*    Mh = (float*)(ws + (size_t)V_DIM * D_DIM * 2
                             + (size_t)M_DIM * D_DIM * 2);            // B*S f32

  const size_t wq = ((size_t)V_DIM * D_DIM) / 4;
  prep_w<<<dim3((unsigned)((wq + 255) / 256)), 256, 0, stream>>>(proj_w, Wh);
  prep_x<<<M_DIM, 256, 0, stream>>>(x, norm_a, norm_b, Xn);
  mh_kernel<<<B_DIM * S_DIM, 256, 0, stream>>>(memory, wh_w, Mh);
  logits_kernel<<<M_DIM, 256, 0, stream>>>(attn, Mh, x, x_t, ws_w, wx_w,
                                           wh_b, ws_b, wx_b, bptr, pgen);
  gemm_wmma<<<dim3((V_DIM + NTILE - 1) / NTILE, M_DIM / 32), 256, 0, stream>>>(
      Xn, Wh, proj_b, out);
  finalize_kernel<<<M_DIM, 256, V_DIM * sizeof(float), stream>>>(
      attn, src, out, pgen);
}
